// DA_RNN_72284299592171
// MI455X (gfx1250) — matrix operand order repaired and checked by reading, not compiled, not run on hardware
//
#include <hip/hip_runtime.h>
#include <hip/hip_bf16.h>

// Problem constants (reference: B=128, T=32, H=256, K=32)
#define B_   128
#define T_   32
#define H_   256
#define K_   32
#define KB_  (K_ * B_)   // 4096 particle rows
#define G4H  (4 * H_)    // 1024 gate columns

typedef __attribute__((ext_vector_type(16))) __bf16 v16bf;
typedef __attribute__((ext_vector_type(8)))  __bf16 v8bf;
typedef __attribute__((ext_vector_type(8)))  float  v8f;

#define WMMA_BF16(a, b, c) \
    __builtin_amdgcn_wmma_f32_16x16x32_bf16(false, (a), false, (b), (short)0, (c), false, false)

// ---------- small device helpers ----------
__device__ __forceinline__ __bf16 f2bf(float x) {
    unsigned u = __builtin_bit_cast(unsigned, x);
    unsigned r = (u + 0x7FFFu + ((u >> 16) & 1u)) >> 16;   // round-to-nearest-even
    unsigned short s = (unsigned short)r;
    return __builtin_bit_cast(__bf16, s);
}
__device__ __forceinline__ float sigm(float x) { return 1.0f / (1.0f + expf(-x)); }
__device__ __forceinline__ float waveSum(float v) {
    for (int d = 16; d > 0; d >>= 1) v += __shfl_xor(v, d, 32);
    return v;
}
__device__ __forceinline__ float waveMax(float v) {
    for (int d = 16; d > 0; d >>= 1) v = fmaxf(v, __shfl_xor(v, d, 32));
    return v;
}
__device__ __forceinline__ unsigned pcg_hash(unsigned x) {
    x = x * 747796405u + 2891336453u;
    unsigned w = ((x >> ((x >> 28) + 4u)) ^ x) * 277803737u;
    return (w >> 22) ^ w;
}
__device__ __forceinline__ float u01(unsigned h) {
    return (h >> 8) * (1.0f / 16777216.0f) + (0.5f / 16777216.0f);
}
// A-fragment (16-bit A 16x32, contiguous k order matching B): lane(l16,half) holds
//   elems 0..15 = Arow[k0 + 16*half + e]
__device__ __forceinline__ v16bf ldA(const __bf16* __restrict__ arow, int k0, int half) {
    v8bf a0 = *(const v8bf*)(arow + k0 + 16 * half);
    v8bf a1 = *(const v8bf*)(arow + k0 + 16 * half + 8);
    return __builtin_shufflevector(a0, a1, 0,1,2,3,4,5,6,7,8,9,10,11,12,13,14,15);
}

// ---------- generic bf16 WMMA GEMM: C(MxN) = A(MxK) * Bt(NxK)^T (+bias[n]) ----------
// 2x2 register blocking: one wave owns a 32x32 output macro-tile (4 accumulators);
// each k-step loads 2 A-fragments + 2 B-fragments and issues 4 WMMAs.
__global__ void wmma_gemm_nt(const __bf16* __restrict__ A, int lda,
                             const __bf16* __restrict__ Bt, int ldb,
                             const float* __restrict__ bias,
                             float* __restrict__ C, int ldc,
                             int M, int N, int Kdim)
{
    int wave  = (blockIdx.x * blockDim.x + threadIdx.x) >> 5;
    int lane  = threadIdx.x & 31;
    int tilesN = N >> 5;
    int tiles  = (M >> 5) * tilesN;
    if (wave >= tiles) return;
    int tm = (wave / tilesN) << 5;
    int tn = (wave % tilesN) << 5;
    int half = lane >> 4, l16 = lane & 15;

    const __bf16* arow0 = A  + (size_t)(tm + l16) * lda;
    const __bf16* arow1 = arow0 + (size_t)16 * lda;
    const __bf16* brow0 = Bt + (size_t)(tn + l16) * ldb;
    const __bf16* brow1 = brow0 + (size_t)16 * ldb;

    v8f acc00 = {}, acc01 = {}, acc10 = {}, acc11 = {};
    for (int k0 = 0; k0 < Kdim; k0 += 32) {
        v16bf a0 = ldA(arow0, k0, half);
        v16bf a1 = ldA(arow1, k0, half);
        v16bf b0 = *(const v16bf*)(brow0 + k0 + 16 * half);
        v16bf b1 = *(const v16bf*)(brow1 + k0 + 16 * half);
        if (k0 + 32 < Kdim) {
            __builtin_prefetch(arow0 + k0 + 64, 0, 0);   // global_prefetch_b8
            __builtin_prefetch(brow0 + k0 + 64, 0, 0);
        }
        acc00 = WMMA_BF16(a0, b0, acc00);
        acc01 = WMMA_BF16(a0, b1, acc01);
        acc10 = WMMA_BF16(a1, b0, acc10);
        acc11 = WMMA_BF16(a1, b1, acc11);
    }
    int n0 = tn + l16, n1 = n0 + 16;
    float bb0 = bias ? bias[n0] : 0.0f;
    float bb1 = bias ? bias[n1] : 0.0f;
    for (int j = 0; j < 8; ++j) {
        int r0 = tm + j + 8 * half, r1 = r0 + 16;
        C[(size_t)r0 * ldc + n0] = acc00[j] + bb0;
        C[(size_t)r0 * ldc + n1] = acc01[j] + bb1;
        C[(size_t)r1 * ldc + n0] = acc10[j] + bb0;
        C[(size_t)r1 * ldc + n1] = acc11[j] + bb1;
    }
}

// ---------- LSTM gates GEMM: TDM-staged B panel in LDS + 2x2 WMMA blocking ----------
// gates(KB x 4H) = h_bf16(KB x H) @ Whh_bf16(4H x H)^T + b_ih[n] + b_hh[n] + yt[row%B]*W_ih[n]
// Block = 8 waves arranged 4(M) x 2(N) of 32x32 wave-tiles -> 128 x 64 block tile.
// The 64-row x 256-col bf16 W_hh panel (32 KB, contiguous) is DMA'd into LDS once per
// block by the Tensor Data Mover, then B fragments are served by ds_load.
#if __has_builtin(__builtin_amdgcn_tensor_load_to_lds)
typedef __attribute__((ext_vector_type(4))) unsigned tdm_u32x4;
typedef __attribute__((ext_vector_type(8))) int      tdm_i32x8;
typedef __attribute__((ext_vector_type(4))) int      tdm_i32x4;
#endif

__global__ void lstm_gates_wmma(const __bf16* __restrict__ A,
                                const __bf16* __restrict__ Whh,
                                const float* __restrict__ b_ih,
                                const float* __restrict__ b_hh,
                                const float* __restrict__ W_ih,
                                const float* __restrict__ yt,
                                float* __restrict__ C)
{
    __shared__ __attribute__((aligned(32))) __bf16 tileB[64 * H_];   // 32 KB

    int bm = blockIdx.x >> 4;          // 0..31  (M blocks of 128)
    int bn = blockIdx.x & 15;          // 0..15  (N blocks of 64)
    const __bf16* gsrc = Whh + (size_t)bn * 64 * H_;   // contiguous 64x256 panel

#if __has_builtin(__builtin_amdgcn_tensor_load_to_lds)
    if ((threadIdx.x >> 5) == 0) {
        // Tensor DMA descriptor (ISA 8.3/8.4): count=1, type=2(image), data_size=2B,
        // tile 256(k) x 64(n), tensor_dim0_stride=256, dest = tileB.
        unsigned lds_off = (unsigned)(size_t)&tileB[0];          // LDS byte address of tileB
        unsigned long long ga = (unsigned long long)(size_t)gsrc;
        tdm_u32x4 g0;
        g0[0] = 1u;                                              // count=1 (valid), user mode
        g0[1] = lds_off;                                         // lds_addr (bytes)
        g0[2] = (unsigned)(ga & 0xFFFFFFFFu);                    // global_addr[31:0]
        g0[3] = (unsigned)((ga >> 32) & 0x1FFFFFFu) | (2u << 30);// global_addr[56:32] | type=2
        tdm_i32x8 g1;
        g1[0] = (1 << 16);                 // workgroup_mask=0, data_size=1 (2 bytes)
        g1[1] = (int)(256u << 16);         // tensor_dim0 = 256 (bits 79:48)
        g1[2] = (int)(1024u << 16);        // tensor_dim1 = 1024 (bits 111:80)
        g1[3] = (int)(256u << 16);         // tile_dim0 = 256 (bits 127:112)
        g1[4] = 64;                        // tile_dim1 = 64, tile_dim2 = 0
        g1[5] = 256;                       // tensor_dim0_stride = 256
        g1[6] = 0;                         // stride0 hi / tensor_dim1_stride lo
        g1[7] = 0;
        tdm_i32x4 gz4 = {0, 0, 0, 0};
        tdm_i32x8 gz8 = {0, 0, 0, 0, 0, 0, 0, 0};
        // 6-arg form (clang-23 / therock-10.0 headers): groups 0..3, extra group, cpol
        __builtin_amdgcn_tensor_load_to_lds(g0, g1, gz4, gz4, gz8, 0);
        __builtin_amdgcn_s_wait_tensorcnt(0);
    }
#else
    // Fallback: cooperative vectorized copy (panel is contiguous 32 KB).
    {
        const uint4* s = (const uint4*)gsrc;
        uint4* d = (uint4*)tileB;
        for (int i = threadIdx.x; i < (64 * H_) / 8; i += 256) d[i] = s[i];
    }
#endif
    __syncthreads();

    int w    = threadIdx.x >> 5;       // 0..7
    int lane = threadIdx.x & 31;
    int wm = w >> 1, wn = w & 1;
    int half = lane >> 4, l16 = lane & 15;
    int tm = bm * 128 + wm * 32;
    int tn_local = wn * 32;
    int tn = bn * 64 + tn_local;

    const __bf16* arow0 = A + (size_t)(tm + l16) * H_;
    const __bf16* arow1 = arow0 + (size_t)16 * H_;
    const __bf16* lb0 = tileB + (size_t)(tn_local + l16) * H_;
    const __bf16* lb1 = lb0 + (size_t)16 * H_;

    v8f acc00 = {}, acc01 = {}, acc10 = {}, acc11 = {};
    for (int k0 = 0; k0 < H_; k0 += 32) {
        v16bf a0 = ldA(arow0, k0, half);
        v16bf a1 = ldA(arow1, k0, half);
        v16bf b0 = *(const v16bf*)(lb0 + k0 + 16 * half);   // ds_load_b128 x2
        v16bf b1 = *(const v16bf*)(lb1 + k0 + 16 * half);
        if (k0 + 32 < H_) {
            __builtin_prefetch(arow0 + k0 + 64, 0, 0);
            __builtin_prefetch(arow1 + k0 + 64, 0, 0);
        }
        acc00 = WMMA_BF16(a0, b0, acc00);
        acc01 = WMMA_BF16(a0, b1, acc01);
        acc10 = WMMA_BF16(a1, b0, acc10);
        acc11 = WMMA_BF16(a1, b1, acc11);
    }
    int n0 = tn + l16, n1 = n0 + 16;
    float bs0 = b_ih[n0] + b_hh[n0], wih0 = W_ih[n0];
    float bs1 = b_ih[n1] + b_hh[n1], wih1 = W_ih[n1];
    for (int j = 0; j < 8; ++j) {
        int r0 = tm + j + 8 * half, r1 = r0 + 16;
        float y0 = yt[r0 & (B_ - 1)], y1 = yt[r1 & (B_ - 1)];
        C[(size_t)r0 * G4H + n0] = acc00[j] + bs0 + y0 * wih0;
        C[(size_t)r0 * G4H + n1] = acc01[j] + bs1 + y0 * wih1;
        C[(size_t)r1 * G4H + n0] = acc10[j] + bs0 + y1 * wih0;
        C[(size_t)r1 * G4H + n1] = acc11[j] + bs1 + y1 * wih1;
    }
}

// ---------- elementwise / reduction kernels ----------
__global__ void init_state(float* hid, float* cel, __bf16* hbf) {
    int id = blockIdx.x * blockDim.x + threadIdx.x;
    if (id >= KB_ * H_) return;
    hid[id] = 0.0f; cel[id] = 0.0f; hbf[id] = f2bf(0.0f);
}

__global__ void f32_to_bf16(const float* __restrict__ s, __bf16* __restrict__ d, int n) {
    int id = blockIdx.x * blockDim.x + threadIdx.x;
    if (id < n) d[id] = f2bf(s[id]);
}

__global__ void mean_particles(const float* __restrict__ hid, const float* __restrict__ cel,
                               float* __restrict__ hm, float* __restrict__ cm,
                               __bf16* __restrict__ hmcm)
{
    int id = blockIdx.x * blockDim.x + threadIdx.x;   // b*H + h
    if (id >= B_ * H_) return;
    int b = id >> 8, h = id & (H_ - 1);
    float sh = 0.0f, sc = 0.0f;
    for (int k = 0; k < K_; ++k) {
        int off = (k * B_ + b) * H_ + h;
        sh += hid[off]; sc += cel[off];
    }
    sh *= (1.0f / K_); sc *= (1.0f / K_);
    hm[id] = sh; cm[id] = sc;
    hmcm[b * (2 * H_) + h]      = f2bf(sh);
    hmcm[b * (2 * H_) + H_ + h] = f2bf(sc);
}

__global__ void attn_score(const float* __restrict__ hmcm_proj,
                           const float* __restrict__ enc_proj,
                           const float* __restrict__ W2, const float* __restrict__ b2,
                           float* __restrict__ a)
{
    int wave = (blockIdx.x * blockDim.x + threadIdx.x) >> 5;
    int lane = threadIdx.x & 31;
    if (wave >= B_ * T_) return;
    int b = wave >> 5;  // T_ == 32
    float s = 0.0f;
    for (int h = lane; h < H_; h += 32)
        s += tanhf(hmcm_proj[b * H_ + h] + enc_proj[wave * H_ + h]) * W2[h];
    s = waveSum(s);
    if (lane == 0) a[wave] = s + b2[0];
}

__global__ void softmax_T(const float* __restrict__ a, float* __restrict__ beta) {
    int wave = (blockIdx.x * blockDim.x + threadIdx.x) >> 5;
    int lane = threadIdx.x & 31;
    if (wave >= B_) return;
    float x = a[wave * T_ + lane];
    float m = waveMax(x);
    float e = expf(x - m);
    float s = waveSum(e);
    beta[wave * T_ + lane] = e / s;
}

__global__ void ctx_kernel(const float* __restrict__ beta, const float* __restrict__ enc,
                           float* __restrict__ ctx)
{
    int id = blockIdx.x * blockDim.x + threadIdx.x;   // b*H + h
    if (id >= B_ * H_) return;
    int b = id >> 8, h = id & (H_ - 1);
    float s = 0.0f;
    for (int t = 0; t < T_; ++t)
        s += beta[b * T_ + t] * enc[(b * T_ + t) * H_ + h];
    ctx[id] = s;
}

__global__ void ytilde_kernel(const float* __restrict__ ctx, const float* __restrict__ fcW,
                              const float* __restrict__ fcb, const float* __restrict__ yprev,
                              int t, float* __restrict__ yt)
{
    int wave = (blockIdx.x * blockDim.x + threadIdx.x) >> 5;
    int lane = threadIdx.x & 31;
    if (wave >= B_) return;
    float s = 0.0f;
    for (int h = lane; h < H_; h += 32) s += ctx[wave * H_ + h] * fcW[h];
    s = waveSum(s);
    if (lane == 0) yt[wave] = s + fcW[H_] * yprev[wave * T_ + t] + fcb[0];
}

__global__ void std_kernel(const float* __restrict__ varW, const float* __restrict__ varb,
                           const float* __restrict__ yt, const float* __restrict__ hm,
                           float* __restrict__ stdv)
{
    int id = blockIdx.x * blockDim.x + threadIdx.x;
    if (id >= B_ * H_) return;
    int b = id >> 8, h = id & (H_ - 1);
    const float* wr = varW + (size_t)h * (H_ + 1);
    float v = wr[0] * yt[b] + varb[h];
    for (int j = 0; j < H_; ++j) v += wr[1 + j] * hm[b * H_ + j];
    stdv[id] = (v > 20.0f) ? v : log1pf(expf(v));
}

__global__ void lstm_pointwise(const float* __restrict__ gates, const float* __restrict__ cells,
                               const float* __restrict__ stdv, int t,
                               float* __restrict__ h_new, float* __restrict__ c_new)
{
    int id = blockIdx.x * blockDim.x + threadIdx.x;   // kb*H + h
    if (id >= KB_ * H_) return;
    int kb = id >> 8, h = id & (H_ - 1), b = kb & (B_ - 1);
    const float* g = gates + (size_t)kb * G4H;
    float gi = g[h], gf = g[H_ + h], gg = g[2 * H_ + h], go = g[3 * H_ + h];
    float c  = sigm(gf) * cells[id] + sigm(gi) * tanhf(gg);
    float hn = sigm(go) * tanhf(c);
    unsigned s0 = pcg_hash((unsigned)id ^ ((unsigned)(t + 1) * 0x9E3779B9u));
    unsigned s1 = pcg_hash(s0 + 0x85EBCA6Bu);
    float eps = sqrtf(fmaxf(-2.0f * logf(u01(s0)), 0.0f)) * cosf(6.28318530718f * u01(s1));
    h_new[id] = hn + eps * stdv[b * H_ + h];
    c_new[id] = c;
}

__global__ void proj_kernel(const float* __restrict__ hN, const float* __restrict__ W,
                            const float* __restrict__ bb, float* __restrict__ proj)
{
    int wave = (blockIdx.x * blockDim.x + threadIdx.x) >> 5;
    int lane = threadIdx.x & 31;
    if (wave >= KB_) return;
    float s = 0.0f;
    for (int h = lane; h < H_; h += 32) s += hN[(size_t)wave * H_ + h] * W[h];
    s = waveSum(s);
    if (lane == 0) proj[wave] = s + bb[0];
}

__global__ void sort_kernel(const float* __restrict__ proj, int* __restrict__ perm)
{
    int wave = (blockIdx.x * blockDim.x + threadIdx.x) >> 5;
    int lane = threadIdx.x & 31;
    if (wave >= B_) return;
    int b = wave;
    float val = proj[lane * B_ + b];
    int   idx = lane;
    for (int k = 2; k <= 32; k <<= 1) {
        for (int j = k >> 1; j > 0; j >>= 1) {
            int partner = lane ^ j;
            float oval = __shfl_xor(val, j, 32);
            int   oidx = __shfl_xor(idx, j, 32);
            bool up = ((lane & k) == 0);
            bool take = (lane < partner) ? (up ? (oval < val) : (oval > val))
                                         : (up ? (oval > val) : (oval < val));
            if (take) { val = oval; idx = oidx; }
        }
    }
    perm[lane * B_ + b] = idx * B_ + b;
}

__global__ void pdf_kernel(const float* __restrict__ hN, const int* __restrict__ perm,
                           const float* __restrict__ pdfW, const float* __restrict__ pdfb,
                           const float* __restrict__ yt, float* __restrict__ out)
{
    int wave = (blockIdx.x * blockDim.x + threadIdx.x) >> 5;
    int lane = threadIdx.x & 31;
    if (wave >= KB_) return;
    int src = perm[wave];
    float s = 0.0f;
    for (int h = lane; h < H_; h += 32) s += hN[(size_t)src * H_ + h] * pdfW[h];
    s = waveSum(s);
    if (lane == 0) out[wave] = s + pdfW[H_] * yt[wave & (B_ - 1)] + pdfb[0];
}

__global__ void resample_kernel(const float* __restrict__ pdfp, const int* __restrict__ perm,
                                int t, int* __restrict__ src_h, int* __restrict__ src_c)
{
    int wave = (blockIdx.x * blockDim.x + threadIdx.x) >> 5;
    int lane = threadIdx.x & 31;
    if (wave >= B_) return;
    int b = wave;
    float w = expf(pdfp[lane * B_ + b]);
    float tot = waveSum(w);
    float cdf = w / tot;
    for (int d = 1; d < 32; d <<= 1) {
        float nv = __shfl_up(cdf, d, 32);
        if (lane >= d) cdf += nv;
    }
    unsigned sd = pcg_hash(((unsigned)(t + 1) * 0xC2B2AE35u) ^ (unsigned)(b * 32 + lane) ^ 0x5bd1e995u);
    float u = u01(sd);
    int samp = 31;
    for (int j = 31; j >= 0; --j) {
        float cj = __shfl(cdf, j, 32);
        if (u <= cj) samp = j;
    }
    src_h[lane * B_ + b] = perm[samp * B_ + b];
    src_c[lane * B_ + b] = samp * B_ + b;
}

__global__ void gather_kernel(const float* __restrict__ hN, const float* __restrict__ cN,
                              const int* __restrict__ src_h, const int* __restrict__ src_c,
                              float* __restrict__ hid, float* __restrict__ cel,
                              __bf16* __restrict__ hbf)
{
    int id = blockIdx.x * blockDim.x + threadIdx.x;
    if (id >= KB_ * H_) return;
    int kb = id >> 8, h = id & (H_ - 1);
    float hv = hN[(size_t)src_h[kb] * H_ + h];
    hid[id] = hv;
    hbf[id] = f2bf(hv);
    cel[id] = cN[(size_t)src_c[kb] * H_ + h];
}

__global__ void ypred_kernel(const float* __restrict__ hm, const float* __restrict__ ctx,
                             const float* __restrict__ decW, const float* __restrict__ decb,
                             const float* __restrict__ encW, const float* __restrict__ encb,
                             float* __restrict__ out)
{
    int wave = (blockIdx.x * blockDim.x + threadIdx.x) >> 5;
    int lane = threadIdx.x & 31;
    if (wave >= B_) return;
    float s = 0.0f;
    for (int h = lane; h < H_; h += 32)
        s += hm[wave * H_ + h] * decW[h] + ctx[wave * H_ + h] * encW[h];
    s = waveSum(s);
    if (lane == 0) out[wave] = s + decb[0] + encb[0];
}

// ---------- host orchestration ----------
extern "C" void kernel_launch(void* const* d_in, const int* in_sizes, int n_in,
                              void* d_out, int out_size, void* d_ws, size_t ws_size,
                              hipStream_t stream)
{
    (void)in_sizes; (void)n_in; (void)out_size; (void)ws_size;
    const float* input_encoded = (const float*)d_in[0];
    const float* y_prev   = (const float*)d_in[1];
    const float* attn_W1  = (const float*)d_in[2];
    const float* attn_b1  = (const float*)d_in[3];
    const float* attn_W2  = (const float*)d_in[4];
    const float* attn_b2  = (const float*)d_in[5];
    const float* fc_W     = (const float*)d_in[6];
    const float* fc_b     = (const float*)d_in[7];
    const float* var_W    = (const float*)d_in[8];
    const float* var_b    = (const float*)d_in[9];
    const float* W_ih     = (const float*)d_in[10];
    const float* W_hh     = (const float*)d_in[11];
    const float* b_ih     = (const float*)d_in[12];
    const float* b_hh     = (const float*)d_in[13];
    const float* fc_dec_W = (const float*)d_in[14];
    const float* fc_dec_b = (const float*)d_in[15];
    const float* pdf_W    = (const float*)d_in[16];
    const float* pdf_b    = (const float*)d_in[17];
    const float* fc_enc_W = (const float*)d_in[18];
    const float* fc_enc_b = (const float*)d_in[19];
    float* out = (float*)d_out;

    char* ws = (char*)d_ws;
    size_t off = 0;
    auto alloc = [&](size_t bytes) -> void* {
        void* p = ws + off;
        off = (off + bytes + 255) & ~(size_t)255;
        return p;
    };
    float*  hiddens   = (float*)alloc(sizeof(float) * KB_ * H_);
    float*  cells     = (float*)alloc(sizeof(float) * KB_ * H_);
    float*  h_new     = (float*)alloc(sizeof(float) * KB_ * H_);
    float*  c_new     = (float*)alloc(sizeof(float) * KB_ * H_);
    float*  gates     = (float*)alloc(sizeof(float) * KB_ * G4H);
    float*  enc_proj  = (float*)alloc(sizeof(float) * B_ * T_ * H_);
    float*  hmcm_proj = (float*)alloc(sizeof(float) * B_ * H_);
    float*  hm        = (float*)alloc(sizeof(float) * B_ * H_);
    float*  cm        = (float*)alloc(sizeof(float) * B_ * H_);
    float*  ctx       = (float*)alloc(sizeof(float) * B_ * H_);
    float*  stdv      = (float*)alloc(sizeof(float) * B_ * H_);
    float*  a_buf     = (float*)alloc(sizeof(float) * B_ * T_);
    float*  beta      = (float*)alloc(sizeof(float) * B_ * T_);
    float*  yt        = (float*)alloc(sizeof(float) * B_);
    float*  proj      = (float*)alloc(sizeof(float) * KB_);
    float*  pdfp      = (float*)alloc(sizeof(float) * KB_);
    int*    perm      = (int*)alloc(sizeof(int) * KB_);
    int*    srch      = (int*)alloc(sizeof(int) * KB_);
    int*    srcc      = (int*)alloc(sizeof(int) * KB_);
    __bf16* hbf       = (__bf16*)alloc(sizeof(__bf16) * KB_ * H_);
    __bf16* encbf     = (__bf16*)alloc(sizeof(__bf16) * B_ * T_ * H_);
    __bf16* w1bf      = (__bf16*)alloc(sizeof(__bf16) * H_ * 3 * H_);
    __bf16* whhbf     = (__bf16*)alloc(sizeof(__bf16) * G4H * H_);
    __bf16* hmcmbf    = (__bf16*)alloc(sizeof(__bf16) * B_ * 2 * H_);
    // total ~42 MiB

    dim3 blk(256);

    // ---- one-time prep ----
    init_state<<<(KB_ * H_) / 256, blk, 0, stream>>>(hiddens, cells, hbf);
    f32_to_bf16<<<(B_ * T_ * H_) / 256, blk, 0, stream>>>(input_encoded, encbf, B_ * T_ * H_);
    f32_to_bf16<<<(H_ * 3 * H_) / 256, blk, 0, stream>>>(attn_W1, w1bf, H_ * 3 * H_);
    f32_to_bf16<<<(G4H * H_) / 256, blk, 0, stream>>>(W_hh, whhbf, G4H * H_);
    // enc_proj = input_encoded @ W1_e.T + b1  (time-invariant -> hoisted out of the scan)
    {
        int tiles = ((B_ * T_) / 32) * (H_ / 32);   // 1024 waves (32x32 macro-tiles)
        wmma_gemm_nt<<<(tiles * 32) / 256, blk, 0, stream>>>(
            encbf, H_, w1bf + 2 * H_, 3 * H_, attn_b1, enc_proj, H_, B_ * T_, H_, H_);
    }

    // ---- sequential scan over T steps ----
    for (int t = 0; t < T_; ++t) {
        mean_particles<<<(B_ * H_) / 256, blk, 0, stream>>>(hiddens, cells, hm, cm, hmcmbf);
        {   // hmcm_proj = [hm|cm] @ W1_{h,c}.T   (M=128, N=256, K=512)
            int tiles = (B_ / 32) * (H_ / 32);      // 32 waves
            wmma_gemm_nt<<<(tiles * 32) / 256, blk, 0, stream>>>(
                hmcmbf, 2 * H_, w1bf, 3 * H_, nullptr, hmcm_proj, H_, B_, H_, 2 * H_);
        }
        attn_score<<<(B_ * T_ * 32) / 256, blk, 0, stream>>>(hmcm_proj, enc_proj, attn_W2, attn_b2, a_buf);
        softmax_T<<<(B_ * 32) / 256, blk, 0, stream>>>(a_buf, beta);
        ctx_kernel<<<(B_ * H_) / 256, blk, 0, stream>>>(beta, input_encoded, ctx);
        ytilde_kernel<<<(B_ * 32) / 256, blk, 0, stream>>>(ctx, fc_W, fc_b, y_prev, t, yt);
        std_kernel<<<(B_ * H_) / 256, blk, 0, stream>>>(var_W, var_b, yt, hm, stdv);
        // gates: grid = 32 (M/128) x 16 (N/64) blocks, TDM-staged B panel per block
        lstm_gates_wmma<<<32 * 16, blk, 0, stream>>>(hbf, whhbf, b_ih, b_hh, W_ih, yt, gates);
        lstm_pointwise<<<(KB_ * H_) / 256, blk, 0, stream>>>(gates, cells, stdv, t, h_new, c_new);
        proj_kernel<<<(KB_ * 32) / 256, blk, 0, stream>>>(h_new, fc_dec_W, fc_dec_b, proj);
        sort_kernel<<<(B_ * 32) / 256, blk, 0, stream>>>(proj, perm);
        pdf_kernel<<<(KB_ * 32) / 256, blk, 0, stream>>>(h_new, perm, pdf_W, pdf_b, yt, pdfp);
        resample_kernel<<<(B_ * 32) / 256, blk, 0, stream>>>(pdfp, perm, t, srch, srcc);
        gather_kernel<<<(KB_ * H_) / 256, blk, 0, stream>>>(h_new, c_new, srch, srcc,
                                                            hiddens, cells, hbf);
    }

    // ---- final prediction ----
    mean_particles<<<(B_ * H_) / 256, blk, 0, stream>>>(hiddens, cells, hm, cm, hmcmbf);
    ypred_kernel<<<(B_ * 32) / 256, blk, 0, stream>>>(hm, ctx, fc_dec_W, fc_dec_b,
                                                      fc_enc_W, fc_enc_b, out);
}
